// ResNeXtBlock_73555609911911
// MI455X (gfx1250) — matrix operand order repaired and checked by reading, not compiled, not run on hardware
//
#include <hip/hip_runtime.h>
#include <hip/hip_bf16.h>
#include <stdint.h>

#define N_VOX   65536
#define C_IN    128
#define C_PATH  32
#define P_PATHS 4
#define K_OFF   27
#define BN_EPS  1e-5f

typedef __bf16 bf16_t;
typedef __attribute__((ext_vector_type(16))) __bf16 v16bf;
typedef __attribute__((ext_vector_type(8)))  __bf16 v8bf;
typedef __attribute__((ext_vector_type(2)))  __bf16 v2bf;
typedef __attribute__((ext_vector_type(8)))  float  v8f;

union V16BF { v16bf v; v8bf h[2]; };
union V8BF  { v8bf v; v2bf p[4]; };

__device__ __forceinline__ bf16_t f2bf(float f) {
  uint32_t u = __builtin_bit_cast(uint32_t, f);
  u += 0x7FFFu + ((u >> 16) & 1u);            // round-to-nearest-even
  return __builtin_bit_cast(bf16_t, (unsigned short)(u >> 16));
}

// Packed fp32x2 -> bf16x2: hardware v_cvt_pk_bf16_f32 when the toolchain
// exposes it for gfx1250, else manual RNE.
__device__ __forceinline__ v2bf pk_bf16(float a, float b) {
#if __has_builtin(__builtin_amdgcn_cvt_pk_bf16_f32)
  return __builtin_bit_cast(v2bf, __builtin_amdgcn_cvt_pk_bf16_f32(a, b));
#else
  v2bf r; r[0] = f2bf(a); r[1] = f2bf(b); return r;
#endif
}

__device__ __forceinline__ void atomic_add_f32(float* p, float v) {
  __hip_atomic_fetch_add(p, v, __ATOMIC_RELAXED, __HIP_MEMORY_SCOPE_AGENT);
}

// Async global->LDS copy of one 16B chunk (GLOBAL_LOAD_ASYNC_TO_LDS_B128,
// tracked by ASYNCcnt). lds_off is the addrspace(3) byte offset.
__device__ __forceinline__ void async_copy_b128(unsigned lds_off, const void* gaddr) {
  asm volatile("global_load_async_to_lds_b128 %0, %1, off"
               :: "v"(lds_off), "v"(gaddr) : "memory");
}
__device__ __forceinline__ void async_wait0() {
  asm volatile("s_wait_asynccnt 0" ::: "memory");
}

// ---------------------------------------------------------------------------
// Packed B layouts (bf16). B tile 32x16: lane = kh*16+n holds
// K = kh*16..kh*16+15 of column n.
// linB: fused 128x128 (4 paths' columns concatenated), same shape as finB.
// ---------------------------------------------------------------------------
#define LIN_E  (4*8*32*16)             // 16384  (fused 128x128)
#define CONV_E (P_PATHS*K_OFF*2*32*16) // 110592
#define FIN_E  (4*8*32*16)             // 16384

__global__ void __launch_bounds__(256)
prep_weights(const float* __restrict__ lin_W, const float* __restrict__ conv_W,
             const float* __restrict__ fin_W,
             bf16_t* __restrict__ linB, bf16_t* __restrict__ convB,
             bf16_t* __restrict__ finB) {
  int gid = blockIdx.x * blockDim.x + threadIdx.x;
  if (gid < LIN_E) {
    int e = gid & 15, lane = (gid >> 4) & 31, nt = (gid >> 9) & 7, kt = gid >> 12;
    int kh = lane >> 4, n = lane & 15;
    int Krow = kt * 32 + kh * 16 + e;
    int col = nt * 16 + n;                 // fused column: p = col>>5, c = col&31
    int p = col >> 5, c = col & 31;
    linB[gid] = f2bf(lin_W[((size_t)p * C_IN + Krow) * C_PATH + c]);
  } else if (gid < LIN_E + CONV_E) {
    int t = gid - LIN_E;
    int e = t & 15, lane = (t >> 4) & 31, nt = (t >> 9) & 1, pk = t >> 10;
    int kh = lane >> 4, n = lane & 15;
    int krow = kh * 16 + e;
    convB[t] = f2bf(conv_W[((size_t)pk * C_PATH + krow) * C_PATH + nt * 16 + n]);
  } else if (gid < LIN_E + CONV_E + FIN_E) {
    int t = gid - LIN_E - CONV_E;
    int e = t & 15, lane = (t >> 4) & 31, nt = (t >> 9) & 7, kt = t >> 12;
    int kh = lane >> 4, n = lane & 15;
    finB[t] = f2bf(fin_W[(size_t)(kt * 32 + kh * 16 + e) * C_IN + nt * 16 + n]);
  }
}

// ---------------------------------------------------------------------------
// One-shot fp32 -> bf16 of feats (each element converted exactly once).
// ---------------------------------------------------------------------------
__global__ void __launch_bounds__(256)
feats_to_bf16(const float* __restrict__ feats, bf16_t* __restrict__ fbf) {
  const size_t i = ((size_t)blockIdx.x * 256 + threadIdx.x) * 8;
  v8f v = *(const v8f*)(feats + i);
  V8BF o;
#pragma unroll
  for (int j = 0; j < 4; ++j) o.p[j] = pk_bf16(v[2 * j], v[2 * j + 1]);
  *(v8bf*)(fbf + i) = o.v;
}

// ---------------------------------------------------------------------------
// Fused path linears: h_all[N,128] = feats_bf @ cat(lin_W) + cat(lin_b).
// 32KB packed B async-staged to LDS per block; 32 WMMAs per wave.
// Epilogue scatters columns into per-path hbf[p][n][32] bf16 rows.
// ---------------------------------------------------------------------------
__global__ void __launch_bounds__(256)
lin_gemm(const bf16_t* __restrict__ fbf, const bf16_t* __restrict__ linB,
         const float* __restrict__ lin_b, bf16_t* __restrict__ hbf) {
  __shared__ __align__(16) bf16_t lsh[LIN_E];        // 32KB of 320KB LDS
  const int tid = threadIdx.x;
  const int lane = tid & 31;
  const int wave = tid >> 5;

#pragma unroll
  for (int i = 0; i < LIN_E / 8 / 256; ++i) {
    const int chunk = i * 256 + tid;
    async_copy_b128((unsigned)(uintptr_t)(&lsh[0]) + (unsigned)chunk * 16u,
                    (const char*)linB + (size_t)chunk * 16);
  }
  async_wait0();
  __syncthreads();

  const int tb = (blockIdx.x * 8 + wave) * 16;
  const int n = lane & 15, kh = lane >> 4;
  const int row = tb + n;

  v8f c[8] = {};
#pragma unroll
  for (int kt = 0; kt < 4; ++kt) {
    const bf16_t* ap = fbf + (size_t)row * C_IN + kt * 32 + kh * 16;
    V16BF a;
    a.h[0] = *(const v8bf*)ap;
    a.h[1] = *(const v8bf*)(ap + 8);
#pragma unroll
    for (int nt = 0; nt < 8; ++nt) {
      v16bf b = *(const v16bf*)(lsh + ((kt * 8 + nt) * 32 + lane) * 16);
      c[nt] = __builtin_amdgcn_wmma_f32_16x16x32_bf16(false, a.v, false, b, (short)0, c[nt], false, false);
    }
  }
#pragma unroll
  for (int nt = 0; nt < 8; ++nt) {
    const int col = nt * 16 + n;                     // fused column
    const int p = col >> 5, cc = col & 31;
    const float bias = lin_b[col];                   // lin_b flat [P*32] == [col]
#pragma unroll
    for (int m = 0; m < 8; ++m) {
      const int r = tb + kh * 8 + m;
      hbf[((size_t)p * N_VOX + r) * C_PATH + cc] = f2bf(c[nt][m] + bias);
    }
  }
}

// ---------------------------------------------------------------------------
// Sparse conv: per (p, k, tile of 16 map entries): gather -> 2 WMMA ->
// scatter-add (fp32 agent atomics into L2-resident accumulators).
// 2KB weight tile async-staged into LDS once per block. Padding is a suffix
// of each map: block-uniform early-out before the barrier, wave-level after.
// ---------------------------------------------------------------------------
__global__ void __launch_bounds__(256)
conv_gemm(const bf16_t* __restrict__ hbf, const bf16_t* __restrict__ convB,
          const int* __restrict__ in_map, const int* __restrict__ out_map,
          float* __restrict__ acc) {
  __shared__ __align__(16) bf16_t bsh[2 * 32 * 16];   // 2KB B tile
  const int tid = threadIdx.x;
  const int lane = tid & 31;
  const int wave = tid >> 5;
  const int k = blockIdx.y;
  const int p = blockIdx.z;
  const size_t mapbase = ((size_t)p * K_OFF + k) * N_VOX;
  const int block_tb = blockIdx.x * (8 * 16);

  if (in_map[mapbase + block_tb] == N_VOX) return;    // block-uniform early out

  if (tid < 128) {
    async_copy_b128((unsigned)(uintptr_t)(&bsh[0]) + (unsigned)tid * 16u,
                    (const char*)(convB + ((size_t)p * K_OFF + k) * (2 * 32 * 16)) + tid * 16);
  }
  async_wait0();
  __syncthreads();

  const int tb = block_tb + wave * 16;
  if (in_map[mapbase + tb] == N_VOX) return;          // wave-level early out
  if (tb + 16 * 8 < N_VOX)
    __builtin_prefetch(&in_map[mapbase + tb + 16 * 8], 0, 1);

  const int n = lane & 15, kh = lane >> 4;
  V16BF a{};
  const int arow = in_map[mapbase + tb + n];
  if (arow < N_VOX) {
    const bf16_t* src = hbf + ((size_t)p * N_VOX + arow) * C_PATH;
    a.h[0] = *(const v8bf*)(src + kh * 16);
    a.h[1] = *(const v8bf*)(src + kh * 16 + 8);
  }
  v16bf b0 = *(const v16bf*)(bsh + lane * 16);        // ds_load_b128 x2
  v16bf b1 = *(const v16bf*)(bsh + (32 + lane) * 16);
  v8f c0 = {}; v8f c1 = {};
  c0 = __builtin_amdgcn_wmma_f32_16x16x32_bf16(false, a.v, false, b0, (short)0, c0, false, false);
  c1 = __builtin_amdgcn_wmma_f32_16x16x32_bf16(false, a.v, false, b1, (short)0, c1, false, false);
#pragma unroll
  for (int m = 0; m < 8; ++m) {
    const int orow = out_map[mapbase + tb + kh * 8 + m];
    if (orow < N_VOX) {
      float* o = acc + ((size_t)p * N_VOX + orow) * C_PATH;
      atomic_add_f32(o + n,      c0[m]);
      atomic_add_f32(o + 16 + n, c1[m]);
    }
  }
}

// ---------------------------------------------------------------------------
// BN training stats: per-(path, channel) sum / sumsq.
// ---------------------------------------------------------------------------
__global__ void __launch_bounds__(256)
bn_stats(const float* __restrict__ acc, float* __restrict__ stats) {
  const int p = blockIdx.x >> 3;
  const int seg = blockIdx.x & 7;
  const int c = threadIdx.x & 31;
  const int g = threadIdx.x >> 5;
  const int rows = N_VOX / 8;
  float s = 0.f, q = 0.f;
  for (int r = seg * rows + g; r < (seg + 1) * rows; r += 8) {
    float v = acc[((size_t)p * N_VOX + r) * C_PATH + c];
    s += v; q += v * v;
  }
  __shared__ float ls[256], lq[256];
  ls[threadIdx.x] = s; lq[threadIdx.x] = q;
  __syncthreads();
  if (g == 0) {
    for (int gg = 1; gg < 8; ++gg) { s += ls[c + 32 * gg]; q += lq[c + 32 * gg]; }
    atomic_add_f32(&stats[p * 64 + c], s);
    atomic_add_f32(&stats[p * 64 + 32 + c], q);
  }
}

// ---------------------------------------------------------------------------
// BN apply + ReLU (leak=0) + concat -> bf16 [N,128].
// ---------------------------------------------------------------------------
__global__ void __launch_bounds__(256)
bn_apply(const float* __restrict__ acc, const float* __restrict__ stats,
         const float* __restrict__ gamma, const float* __restrict__ beta,
         bf16_t* __restrict__ catbf) {
  int e = blockIdx.x * blockDim.x + threadIdx.x;   // over N*128
  int n = e >> 7;
  int pc = e & 127;
  int p = pc >> 5, c = pc & 31;
  float mu  = stats[p * 64 + c] * (1.f / N_VOX);
  float var = stats[p * 64 + 32 + c] * (1.f / N_VOX) - mu * mu;
  float scale = rsqrtf(var + BN_EPS) * gamma[p * C_PATH + c];
  float shift = beta[p * C_PATH + c] - mu * scale;
  float v = acc[((size_t)p * N_VOX + n) * C_PATH + c] * scale + shift;
  catbf[(size_t)n * C_IN + pc] = f2bf(fmaxf(v, 0.f));
}

// ---------------------------------------------------------------------------
// Final linear + bias + residual: out = cat @ fin_W + fin_b + feats.
// ---------------------------------------------------------------------------
__global__ void __launch_bounds__(256)
fin_gemm(const bf16_t* __restrict__ catbf, const bf16_t* __restrict__ finB,
         const float* __restrict__ fin_b, const float* __restrict__ feats,
         float* __restrict__ out) {
  __shared__ __align__(16) bf16_t fsh[FIN_E];        // 32KB of 320KB LDS
  const int tid = threadIdx.x;
  const int lane = tid & 31;
  const int wave = tid >> 5;

#pragma unroll
  for (int i = 0; i < FIN_E / 8 / 256; ++i) {
    const int chunk = i * 256 + tid;
    async_copy_b128((unsigned)(uintptr_t)(&fsh[0]) + (unsigned)chunk * 16u,
                    (const char*)finB + (size_t)chunk * 16);
  }
  async_wait0();
  __syncthreads();

  const int tb = (blockIdx.x * 8 + wave) * 16;
  const int n = lane & 15, kh = lane >> 4;
  const int row = tb + n;

  v8f c[8] = {};
#pragma unroll
  for (int kt = 0; kt < 4; ++kt) {
    const bf16_t* ap = catbf + (size_t)row * C_IN + kt * 32 + kh * 16;
    V16BF a;
    a.h[0] = *(const v8bf*)ap;
    a.h[1] = *(const v8bf*)(ap + 8);
#pragma unroll
    for (int nt = 0; nt < 8; ++nt) {
      v16bf b = *(const v16bf*)(fsh + ((kt * 8 + nt) * 32 + lane) * 16);
      c[nt] = __builtin_amdgcn_wmma_f32_16x16x32_bf16(false, a.v, false, b, (short)0, c[nt], false, false);
    }
  }
#pragma unroll
  for (int nt = 0; nt < 8; ++nt) {
    const int col = nt * 16 + n;
    const float bias = fin_b[col];
#pragma unroll
    for (int m = 0; m < 8; ++m) {
      const size_t idx = (size_t)(tb + kh * 8 + m) * C_IN + col;
      out[idx] = c[nt][m] + bias + feats[idx];
    }
  }
}

// ---------------------------------------------------------------------------
extern "C" void kernel_launch(void* const* d_in, const int* in_sizes, int n_in,
                              void* d_out, int out_size, void* d_ws, size_t ws_size,
                              hipStream_t stream) {
  (void)in_sizes; (void)n_in; (void)out_size; (void)ws_size;
  const float* feats  = (const float*)d_in[0];
  const float* lin_W  = (const float*)d_in[1];
  const float* lin_b  = (const float*)d_in[2];
  const float* conv_W = (const float*)d_in[3];
  const float* gamma  = (const float*)d_in[4];
  const float* beta   = (const float*)d_in[5];
  const float* fin_W  = (const float*)d_in[6];
  const float* fin_b  = (const float*)d_in[7];
  const int*   in_map = (const int*)d_in[8];
  const int*   out_map= (const int*)d_in[9];
  float* out = (float*)d_out;

  char* ws = (char*)d_ws;
  size_t off = 0;
  auto take = [&](size_t bytes) -> char* {
    char* p = ws + off;
    off = (off + bytes + 255) & ~(size_t)255;
    return p;
  };
  bf16_t* fbf   = (bf16_t*)take((size_t)N_VOX * C_IN * 2);
  bf16_t* hbf   = (bf16_t*)take((size_t)P_PATHS * N_VOX * C_PATH * 2);
  float*  acc   = (float*) take((size_t)P_PATHS * N_VOX * C_PATH * 4);
  float*  stats = (float*) take((size_t)P_PATHS * 64 * 4);
  bf16_t* catbf = (bf16_t*)take((size_t)N_VOX * C_IN * 2);
  bf16_t* linB  = (bf16_t*)take((size_t)LIN_E * 2);
  bf16_t* convB = (bf16_t*)take((size_t)CONV_E * 2);
  bf16_t* finB  = (bf16_t*)take((size_t)FIN_E * 2);

  // Zero conv accumulators + BN stats (contiguous span), graph-capture safe.
  size_t zbytes = (size_t)((char*)stats - (char*)acc) + (size_t)P_PATHS * 64 * 4;
  hipMemsetAsync(acc, 0, zbytes, stream);

  prep_weights<<<(LIN_E + CONV_E + FIN_E + 255) / 256, 256, 0, stream>>>(
      lin_W, conv_W, fin_W, linB, convB, finB);

  feats_to_bf16<<<(N_VOX * C_IN) / 8 / 256, 256, 0, stream>>>(feats, fbf);

  lin_gemm<<<N_VOX / 16 / 8, 256, 0, stream>>>(fbf, linB, lin_b, hbf);

  dim3 cg(N_VOX / 16 / 8, K_OFF, P_PATHS);           // 512 x 27 x 4
  conv_gemm<<<cg, 256, 0, stream>>>(hbf, convB, in_map, out_map, acc);

  bn_stats<<<P_PATHS * 8, 256, 0, stream>>>(acc, stats);
  bn_apply<<<(N_VOX * C_IN) / 256, 256, 0, stream>>>(acc, stats, gamma, beta, catbf);

  fin_gemm<<<N_VOX / 16 / 8, 256, 0, stream>>>(catbf, finB, fin_b, feats, out);
}
